// GroupShuffleAttention_18889266167879
// MI455X (gfx1250) — matrix operand order repaired and checked by reading, not compiled, not run on hardware
//
#include <hip/hip_runtime.h>
#include <hip/hip_bf16.h>
#include <math.h>

// ---- types for WMMA fragments ----
typedef __attribute__((ext_vector_type(16))) _Float16 v16h;
typedef __attribute__((ext_vector_type(8)))  _Float16 h8;
typedef __attribute__((ext_vector_type(4)))  _Float16 h4;
typedef __attribute__((ext_vector_type(8)))  float    v8f;

#define BB   2
#define CC   256
#define NN   4096
#define GG   8
#define CPG  32      // channels per attention group (== WMMA K!)
#define GN_CH 8      // channels per GroupNorm group (32 groups)
#define EPSV 1e-5f

// ---------------------------------------------------------------------------
// Kernel 1: grouped 1x1 conv -> x (f16) and elu(x) (f16) into workspace.
// ~0.13 GFLOP total; plain VALU FMA, weights cached in LDS.
// ---------------------------------------------------------------------------
__global__ __launch_bounds__(256) void conv_elu_kernel(
    const float* __restrict__ pts, const float* __restrict__ w,
    const float* __restrict__ bias, _Float16* __restrict__ xh,
    _Float16* __restrict__ eh)
{
    __shared__ float sW[CPG][CPG];
    __shared__ float sB[CPG];
    int blk   = blockIdx.x;
    int chunk = blk & 15;          // 16 chunks of 256 columns
    int bg    = blk >> 4;          // 0..15  (b*8+g)
    int g     = bg & 7;
    int b     = bg >> 3;
    const float* wg = w + g * CPG * CPG;
    for (int t = threadIdx.x; t < CPG * CPG; t += 256) sW[t / CPG][t % CPG] = wg[t];
    if (threadIdx.x < CPG) sB[threadIdx.x] = bias[g * CPG + threadIdx.x];
    __syncthreads();

    int n = chunk * 256 + threadIdx.x;
    const float* pcol = pts + ((size_t)(b * CC + g * CPG)) * NN + n;
    float p[CPG];
#pragma unroll
    for (int j = 0; j < CPG; ++j) p[j] = pcol[(size_t)j * NN];

    _Float16* xcol = xh + ((size_t)bg * CPG) * NN + n;
    _Float16* ecol = eh + ((size_t)bg * CPG) * NN + n;
#pragma unroll 4
    for (int i = 0; i < CPG; ++i) {
        float acc = sB[i];
#pragma unroll
        for (int j = 0; j < CPG; ++j) acc = fmaf(sW[i][j], p[j], acc);
        xcol[(size_t)i * NN] = (_Float16)acc;
        float e = acc > 0.f ? acc : (__expf(acc) - 1.f);   // ELU
        ecol[(size_t)i * NN] = (_Float16)e;
    }
}

// A-fragment loader: 16-bit A layout (ISA 7.12.2): lanes 0-15 take K halves
// {j..j+7, j+16..}, lanes 16-31 the +8 shifted halves. Rows are 64B-aligned
// so each half-fragment is one 16-byte LDS read.
__device__ inline v16h loadA16(const _Float16* rowptr, int base)
{
    h8 lo = *(const h8*)(rowptr + base);        // K = base .. base+7
    h8 hi = *(const h8*)(rowptr + 16 + base);   // K = base+16 .. base+23
    v16h a;
#pragma unroll
    for (int j = 0; j < 8; ++j) { a[j] = lo[j]; a[j + 8] = hi[j]; }
    return a;
}

// ---------------------------------------------------------------------------
// Kernel 2: flash-style streaming attention, softmax over the contraction
// axis n.  One WG = (b, g, 128-column m tile); 8 waves x 16 columns.
// Per 32-column n step: 2 WMMAs for S (K = c = 32 exactly), online
// max/sum per output column (lane-pair shuffle, wave32), 2 WMMAs for the
// elu(X) @ P update.  Epilogue fuses channel shuffle + identity add.
// ---------------------------------------------------------------------------
__global__ __launch_bounds__(256) void flash_attn_kernel(
    const _Float16* __restrict__ xh, const _Float16* __restrict__ eh,
    const float* __restrict__ pts, float* __restrict__ y)
{
    __shared__ __align__(16) _Float16 sKT[32][32];     // [n_local][i]  (X^T tile)
    __shared__ __align__(16) _Float16 sV [32][32];     // [i][n_local]  (elu tile)
    __shared__ __align__(16) _Float16 sP [8][32][16];  // per-wave P transpose

    int tileM = blockIdx.x & 31;     // 32 m tiles of 128 columns
    int bg    = blockIdx.x >> 5;     // b*8+g
    int g     = bg & 7;
    int b     = bg >> 3;
    int wv    = threadIdx.x >> 5;
    int lane  = threadIdx.x & 31;
    int m0    = tileM * 128 + wv * 16;

    const _Float16* xg = xh + ((size_t)bg * CPG) * NN;
    const _Float16* eg = eh + ((size_t)bg * CPG) * NN;

    // Q fragment (B layout: N = m column = lane%16, elements = K = channel i in table order).
    v16h q;
    {
        const _Float16* qp = xg + m0 + (lane & 15);
        const int kb = (lane >> 4) * 8;
#pragma unroll
        for (int j = 0; j < 8; ++j) { q[j] = qp[(size_t)(kb + j) * NN]; q[j + 8] = qp[(size_t)(kb + 16 + j) * NN]; }
    }

    v8f o0 = {}, o1 = {};
    float mx = -__builtin_inff(), lsum = 0.f;
    // 1/sqrt(32) folded with log2(e): exp(x) == exp2(x*log2e) -> v_exp_f32
    const float sc2 = 0.17677669529663688f * 1.44269504088896340736f;

    const int base = (lane < 16) ? 0 : 8;
    const int Mr   = lane & 15;

    for (int n0 = 0; n0 < NN; n0 += 32) {
        __syncthreads();                       // previous-iter LDS reads done
        {   // cooperative staging of K^T / V tiles (shared by all 8 waves)
            int i   = threadIdx.x >> 3;        // channel 0..31
            int nn0 = (threadIdx.x & 7) * 4;   // n_local 0..28
            h4 kv = *(const h4*)(xg + (size_t)i * NN + n0 + nn0);
            h4 vv = *(const h4*)(eg + (size_t)i * NN + n0 + nn0);
#pragma unroll
            for (int u = 0; u < 4; ++u) sKT[nn0 + u][i] = kv[u];
            *(h4*)&sV[i][nn0] = vv;
            if (n0 + 32 < NN) {                // pull next tile toward L2/L0
                __builtin_prefetch(xg + (size_t)i * NN + n0 + 32 + nn0, 0, 1);
                __builtin_prefetch(eg + (size_t)i * NN + n0 + 32 + nn0, 0, 1);
            }
        }
        __syncthreads();

        // S tiles: A = X^T rows (n), B = Q; one WMMA covers full K = c = 32.
        v8f zc = {};
        v16h a0 = loadA16(&sKT[Mr][0],      base);
        v16h a1 = loadA16(&sKT[16 + Mr][0], base);
        v8f s0 = __builtin_amdgcn_wmma_f32_16x16x32_f16(false, a0, false, q,
                                                        (short)0, zc, false, false);
        v8f s1 = __builtin_amdgcn_wmma_f32_16x16x32_f16(false, a1, false, q,
                                                        (short)0, zc, false, false);

        // online softmax statistics per output column (C/D layout: N = lane%16,
        // lanes l and l^16 hold the two M halves of the same column).
        float ml = -__builtin_inff();
#pragma unroll
        for (int r = 0; r < 8; ++r) {
            s0[r] *= sc2; s1[r] *= sc2;
            ml = fmaxf(ml, fmaxf(s0[r], s1[r]));
        }
        ml = fmaxf(ml, __shfl_xor(ml, 16, 32));
        float nm   = fmaxf(mx, ml);
        float corr = exp2f(mx - nm);           // first iter: exp2(-inf) = 0
        float ps   = 0.f;
#pragma unroll
        for (int r = 0; r < 8; ++r) {
            float p0 = exp2f(s0[r] - nm);
            float p1 = exp2f(s1[r] - nm);
            ps += p0 + p1;
            sP[wv][r + 8 * (lane >> 4)][lane & 15]      = (_Float16)p0;
            sP[wv][16 + r + 8 * (lane >> 4)][lane & 15] = (_Float16)p1;
        }
        ps  += __shfl_xor(ps, 16, 32);
        lsum = lsum * corr + ps;
        mx   = nm;
#pragma unroll
        for (int r = 0; r < 8; ++r) { o0[r] *= corr; o1[r] *= corr; }

        // wave-private LDS transpose: wait our own DS stores, fence scheduler
        asm volatile("s_wait_dscnt 0" ::: "memory");
        __builtin_amdgcn_wave_barrier();

        // P in B layout: N = m column = lane%16, elements = K = n_local in table order.
        v16h pb;
        {
#pragma unroll
            for (int j = 0; j < 8; ++j) { pb[j] = sP[wv][base + j][lane & 15]; pb[j + 8] = sP[wv][16 + base + j][lane & 15]; }
        }
        // elu(X) rows as A operands (two c row-blocks), contract over n (K=32).
        v16h av0 = loadA16(&sV[Mr][0],      base);
        v16h av1 = loadA16(&sV[16 + Mr][0], base);
        o0 = __builtin_amdgcn_wmma_f32_16x16x32_f16(false, av0, false, pb,
                                                    (short)0, o0, false, false);
        o1 = __builtin_amdgcn_wmma_f32_16x16x32_f16(false, av1, false, pb,
                                                    (short)0, o1, false, false);
    }

    // epilogue: divide by softmax denom, channel shuffle (ch = c_local*G + g),
    // add identity, store f32 y for the GroupNorm passes.
    float inv = 1.f / lsum;
    int   col = m0 + (lane & 15);
    const float* pid = pts + (size_t)b * CC * NN;
    float*       yb  = y   + (size_t)b * CC * NN;
#pragma unroll
    for (int r = 0; r < 8; ++r) {
        int c0 = r + 8 * (lane >> 4);
        size_t i0 = (size_t)(c0 * GG + g) * NN + col;
        yb[i0] = o0[r] * inv + pid[i0];
        size_t i1 = (size_t)((c0 + 16) * GG + g) * NN + col;
        yb[i1] = o1[r] * inv + pid[i1];
    }
}

// ---------------------------------------------------------------------------
// Kernel 3: GroupNorm statistics. One block per (b, gn_group): contiguous
// 8x4096 slab -> mean/var, no atomics needed.
// ---------------------------------------------------------------------------
__global__ __launch_bounds__(256) void gn_stats_kernel(
    const float* __restrict__ y, float2* __restrict__ stats)
{
    __shared__ float rs[256];
    __shared__ float rq[256];
    int gi = blockIdx.x;                          // b*32 + group
    const float* p = y + (size_t)gi * GN_CH * NN;
    float s = 0.f, q = 0.f;
    for (int i = threadIdx.x; i < GN_CH * NN; i += 256) {
        float v = p[i]; s += v; q += v * v;
    }
    rs[threadIdx.x] = s; rq[threadIdx.x] = q;
    __syncthreads();
    for (int o = 128; o > 0; o >>= 1) {
        if (threadIdx.x < o) {
            rs[threadIdx.x] += rs[threadIdx.x + o];
            rq[threadIdx.x] += rq[threadIdx.x + o];
        }
        __syncthreads();
    }
    if (threadIdx.x == 0) {
        const float cnt = (float)(GN_CH * NN);
        float mean = rs[0] / cnt;
        float var  = rq[0] / cnt - mean * mean;
        stats[gi]  = make_float2(mean, var);
    }
}

// ---------------------------------------------------------------------------
// Kernel 4: apply GroupNorm affine -> d_out.
// ---------------------------------------------------------------------------
__global__ __launch_bounds__(256) void gn_apply_kernel(
    const float* __restrict__ y, const float2* __restrict__ stats,
    const float* __restrict__ gw, const float* __restrict__ gb,
    float* __restrict__ out)
{
    size_t idx = (size_t)blockIdx.x * 256 + threadIdx.x;
    int ch = (int)((idx / NN) % CC);
    int gi = (int)(idx / ((size_t)GN_CH * NN));   // groups are contiguous slabs
    float2 st = stats[gi];
    float v = (y[idx] - st.x) * rsqrtf(st.y + EPSV);
    out[idx] = v * gw[ch] + gb[ch];
}

// ---------------------------------------------------------------------------
extern "C" void kernel_launch(void* const* d_in, const int* in_sizes, int n_in,
                              void* d_out, int out_size, void* d_ws, size_t ws_size,
                              hipStream_t stream)
{
    const float* pts = (const float*)d_in[0];   // [B,C,N]
    const float* cw  = (const float*)d_in[1];   // [G,c,c]
    const float* cb  = (const float*)d_in[2];   // [C]
    const float* gw  = (const float*)d_in[3];   // [C]
    const float* gb  = (const float*)d_in[4];   // [C]
    float* out = (float*)d_out;

    // workspace layout: x_f16 (4MB) | elu_f16 (4MB) | y f32 (8MB) | stats
    char* ws = (char*)d_ws;
    _Float16* xh   = (_Float16*)ws;
    _Float16* eh   = (_Float16*)(ws + ((size_t)4 << 20));
    float*    y    = (float*)   (ws + ((size_t)8 << 20));
    float2*   stat = (float2*)  (ws + ((size_t)16 << 20));

    conv_elu_kernel  <<<BB * GG * 16, 256, 0, stream>>>(pts, cw, cb, xh, eh);
    flash_attn_kernel<<<BB * GG * 32, 256, 0, stream>>>(xh, eh, pts, y);
    gn_stats_kernel  <<<BB * 32,      256, 0, stream>>>(y, stat);
    gn_apply_kernel  <<<(BB * CC * NN) / 256, 256, 0, stream>>>(y, stat, gw, gb, out);
}
